// FusionCPI_20452634263712
// MI455X (gfx1250) — matrix operand order repaired and checked by reading, not compiled, not run on hardware
//
#include <hip/hip_runtime.h>
#include <hip/hip_bf16.h>

// ---------------------------------------------------------------------------
// Types for CDNA5 WMMA (wave32, 16x16x32 bf16 -> f32 accumulate)
// ---------------------------------------------------------------------------
typedef __bf16 bf16_t;
typedef __attribute__((ext_vector_type(16))) __bf16 v16bf;
typedef __attribute__((ext_vector_type(8)))  __bf16 v8bf;
typedef __attribute__((ext_vector_type(8)))  float  v8f;

__device__ __forceinline__ bf16_t f2bf(float f) {
    unsigned u = __builtin_bit_cast(unsigned, f);
    unsigned r = u + 0x7FFFu + ((u >> 16) & 1u);   // round-to-nearest-even
    unsigned short h = (unsigned short)(r >> 16);
    return __builtin_bit_cast(bf16_t, h);
}

// Epilogues: 0 = identity, 1 = ELU, 2 = ReLU, 3 = LeakyReLU(0.2)
template<int EPI>
__device__ __forceinline__ float epi(float x) {
    if (EPI == 1) return x > 0.f ? x : (__expf(x) - 1.f);
    if (EPI == 2) return x > 0.f ? x : 0.f;
    if (EPI == 3) return x > 0.f ? x : 0.2f * x;
    return x;
}

__device__ __forceinline__ void storeOut(float*  p, float v) { *p = v; }
__device__ __forceinline__ void storeOut(bf16_t* p, float v) { *p = f2bf(v); }

// A fragment: lane holds row (lane&15);
// elements 0..15 = K khalf+0..15 (contiguous, matches B).
__device__ __forceinline__ v16bf loadA(const bf16_t* p) {
    v16bf a = *(const v16bf*)p;
    return a;
}

// ---------------------------------------------------------------------------
// Generic batched WMMA GEMM:  C[z] = epi( A[z] (MxK) * Bt[z]^T (NxK) + bias )
// A row-major bf16 (lda), Bt row-major bf16 (ldb=K stride), C (ldc) f32/bf16.
// 2x2 register-blocked: one 32x32 output tile per wave (4 WMMAs / k-step,
// each A/B fragment reused twice). 4 waves/block cover 128 output columns.
// C batch offset = (z/zDiv)*strideC1 + (z%zDiv)*strideC2 (handles head concat).
// ---------------------------------------------------------------------------
template<int EPI, typename OutT>
__global__ __launch_bounds__(128)
void gemm_bf16_wmma(const bf16_t* __restrict__ A, int lda, long strideA,
                    const bf16_t* __restrict__ Bt, int ldb, long strideBt,
                    OutT* __restrict__ C, int ldc,
                    int zDiv, long strideC1, long strideC2,
                    const float* __restrict__ bias,
                    int M, int N, int K)
{
    const int lane = threadIdx.x & 31;
    const int wave = threadIdx.x >> 5;
    const int tm = blockIdx.x;               // 32-row tile
    const int tn = blockIdx.y * 4 + wave;    // 32-col tile
    if (tn * 32 >= N) return;                // whole-wave exit (EXEC stays all-1)

    const int z = blockIdx.z;
    A  += (long)z * strideA;
    Bt += (long)z * strideBt;
    C  += (long)(z / zDiv) * strideC1 + (long)(z % zDiv) * strideC2;

    const int khalfA = (lane >> 4) * 16;     // A: K 0..15 / 16..31 split per half-wave
    const int khalfB = (lane >> 4) * 16;     // B: K 0..15 / 16..31 split per half-wave
    const bf16_t* aPtr0 = A  + (long)(tm * 32 + (lane & 15)) * lda + khalfA;
    const bf16_t* aPtr1 = aPtr0 + 16L * lda;
    const bf16_t* bPtr0 = Bt + (long)(tn * 32 + (lane & 15)) * ldb + khalfB;
    const bf16_t* bPtr1 = bPtr0 + 16L * ldb;

    v8f acc00 = {}, acc01 = {}, acc10 = {}, acc11 = {};
    for (int k = 0; k < K; k += 32) {
        // Prefetch ~4 k-iterations ahead (speculative; OOB safely dropped).
        __builtin_prefetch(aPtr0 + k + 128, 0, 3);
        __builtin_prefetch(aPtr1 + k + 128, 0, 3);
        __builtin_prefetch(bPtr0 + k + 128, 0, 3);
        __builtin_prefetch(bPtr1 + k + 128, 0, 3);

        v16bf a0 = loadA(aPtr0 + k);
        v16bf a1 = loadA(aPtr1 + k);
        v16bf b0 = *(const v16bf*)(bPtr0 + k);   // 16 contiguous K values
        v16bf b1 = *(const v16bf*)(bPtr1 + k);

        acc00 = __builtin_amdgcn_wmma_f32_16x16x32_bf16(false, a0, false, b0, (short)0, acc00, false, false);
        acc01 = __builtin_amdgcn_wmma_f32_16x16x32_bf16(false, a0, false, b1, (short)0, acc01, false, false);
        acc10 = __builtin_amdgcn_wmma_f32_16x16x32_bf16(false, a1, false, b0, (short)0, acc10, false, false);
        acc11 = __builtin_amdgcn_wmma_f32_16x16x32_bf16(false, a1, false, b1, (short)0, acc11, false, false);
    }

    // C/D layout: lane<16 -> col=lane, rows +0..7; lane>=16 -> rows +8..15.
    const int rbase = (lane >> 4) * 8;
    const int cbase = lane & 15;
#pragma unroll
    for (int j = 0; j < 2; ++j) {
        const int ccol = tn * 32 + j * 16 + cbase;
        const float bv = bias ? bias[ccol] : 0.f;
#pragma unroll
        for (int i = 0; i < 2; ++i) {
            const v8f& acc = (i == 0) ? (j == 0 ? acc00 : acc01)
                                      : (j == 0 ? acc10 : acc11);
            const int crow0 = tm * 32 + i * 16 + rbase;
#pragma unroll
            for (int r = 0; r < 8; ++r) {
                float v = epi<EPI>(acc[r] + bv);
                storeOut(C + (long)(crow0 + r) * ldc + ccol, v);
            }
        }
    }
}

// ---------------------------------------------------------------------------
// Helper kernels
// ---------------------------------------------------------------------------

// av[b*512+n][c] = bf16(emb_atom[atoms[b*512+n]][c])   (COMP = 128)
__global__ void embed_kernel(bf16_t* __restrict__ av, const int* __restrict__ atoms,
                             const float* __restrict__ emb, long total)
{
    long tid = (long)blockIdx.x * blockDim.x + threadIdx.x;
    if (tid >= total) return;
    int n = (int)(tid >> 7);
    int c = (int)(tid & 127);
    av[tid] = f2bf(emb[(long)atoms[n] * 128 + c]);
}

// dst[tid] = bf16(src[tid])
__global__ void cvt_bf16_kernel(bf16_t* __restrict__ dst, const float* __restrict__ src,
                                long total)
{
    long tid = (long)blockIdx.x * blockDim.x + threadIdx.x;
    if (tid >= total) return;
    dst[tid] = f2bf(src[tid]);
}

// Batched transpose+convert: for batch z, src matrix S_z[r][c] (RxCc, lda) at
// offset (z/zDiv)*s1 + (z%zDiv)*s2; dst[z*Cc*R + c*R + r] = bf16(S_z[r][c]).
__global__ void transpose_f32_bf16(bf16_t* __restrict__ dst, const float* __restrict__ src,
                                   int R, int Cc, int lda, int zDiv, long s1, long s2,
                                   long total)
{
    long tid = (long)blockIdx.x * blockDim.x + threadIdx.x;
    if (tid >= total) return;
    long per = (long)R * Cc;
    int  z   = (int)(tid / per);
    long rem = tid - (long)z * per;
    int  c   = (int)(rem / R);
    int  r   = (int)(rem - (long)c * R);
    long srcIdx = (long)(z / zDiv) * s1 + (long)(z % zDiv) * s2 + (long)r * lda + c;
    dst[tid] = f2bf(src[srcIdx]);
}

// GAT head scores: s1[z*512+n] = sum_f Wh[(b*512+n)*256 + h*64 + f] * a_gat[h*128 + f]
//                  s2 with a_gat[h*128 + 64 + f];  z = b*4 + h
__global__ void gat1_scores_kernel(float* __restrict__ s1, float* __restrict__ s2,
                                   const float* __restrict__ Wh,
                                   const float* __restrict__ a_gat)
{
    int tid = blockIdx.x * blockDim.x + threadIdx.x;
    if (tid >= 16 * 4 * 512) return;
    int n = tid & 511;
    int h = (tid >> 9) & 3;
    int b = tid >> 11;
    const float* row = Wh + (long)(b * 512 + n) * 256 + h * 64;
    const float* a1  = a_gat + h * 128;
    float x = 0.f, y = 0.f;
#pragma unroll 8
    for (int f = 0; f < 64; ++f) { float w = row[f]; x += w * a1[f]; y += w * a1[64 + f]; }
    s1[tid] = x;
    s2[tid] = y;
}

// Second GAT scores over Wh2 [8192,128] with a_out [256]
__global__ void gat2_scores_kernel(float* __restrict__ s1, float* __restrict__ s2,
                                   const float* __restrict__ Wh2,
                                   const float* __restrict__ a_out)
{
    int tid = blockIdx.x * blockDim.x + threadIdx.x;
    if (tid >= 8192) return;
    const float* row = Wh2 + (long)tid * 128;
    float x = 0.f, y = 0.f;
#pragma unroll 8
    for (int o = 0; o < 128; ++o) { float w = row[o]; x += w * a_out[o]; y += w * a_out[128 + o]; }
    s1[tid] = x;
    s2[tid] = y;
}

// Masked attention softmax, one wave32 per row of 512 columns.
// e_j = leaky(s1[row] + s2[z*512+j]); masked by adjacency>0 with NEG; softmax;
// normalized probabilities written as bf16 (A-matrix of the att x V GEMM).
__global__ __launch_bounds__(256)
void gat_softmax_kernel(bf16_t* __restrict__ att, const float* __restrict__ s1,
                        const float* __restrict__ s2, const int* __restrict__ adj,
                        int Hdiv, long nrows)
{
    const int lane = threadIdx.x & 31;
    const int wave = threadIdx.x >> 5;
    long rowIdx = (long)blockIdx.x * 8 + wave;
    if (rowIdx >= nrows) return;
    int z = (int)(rowIdx >> 9);
    int n = (int)(rowIdx & 511);
    int b = z / Hdiv;
    const int*   adjRow = adj + (long)(b * 512 + n) * 512;
    const float* s2p    = s2 + (long)z * 512;
    const float  s1v    = s1[rowIdx];

    float e[16];
    float mx = -3.0e38f;
#pragma unroll
    for (int t = 0; t < 16; ++t) {
        int j = lane + 32 * t;
        float v = s1v + s2p[j];
        v = v > 0.f ? v : 0.2f * v;            // leaky_relu(0.2)
        v = adjRow[j] > 0 ? v : -9.0e15f;      // NEG mask
        e[t] = v;
        mx = fmaxf(mx, v);
    }
    for (int m = 16; m; m >>= 1) mx = fmaxf(mx, __shfl_xor(mx, m, 32));
    float sum = 0.f;
#pragma unroll
    for (int t = 0; t < 16; ++t) { e[t] = __expf(e[t] - mx); sum += e[t]; }
    for (int m = 16; m; m >>= 1) sum += __shfl_xor(sum, m, 32);
    float inv = 1.0f / sum;
    bf16_t* arow = att + rowIdx * 512;
#pragma unroll
    for (int t = 0; t < 16; ++t) arow[lane + 32 * t] = f2bf(e[t] * inv);
}

// ---------------------------------------------------------------------------
// Orchestration
// ---------------------------------------------------------------------------
extern "C" void kernel_launch(void* const* d_in, const int* in_sizes, int n_in,
                              void* d_out, int out_size, void* d_ws, size_t ws_size,
                              hipStream_t stream)
{
    const int*   atoms = (const int*)  d_in[0];   // [16,512]
    const int*   adj   = (const int*)  d_in[1];   // [16,512,512]
    const float* prot  = (const float*)d_in[3];   // [16,1024,1152]
    const float* emb   = (const float*)d_in[5];   // [45,128]
    const float* Wgat  = (const float*)d_in[6];   // [4,128,64]
    const float* agat  = (const float*)d_in[7];   // [4,128,1]
    const float* Wout  = (const float*)d_in[8];   // [256,128]
    const float* aout  = (const float*)d_in[9];   // [256,1]
    const float* Wc    = (const float*)d_in[10];  // [128,128]
    const float* bc    = (const float*)d_in[11];  // [128]
    const float* W1    = (const float*)d_in[12];  // [1152,512]
    const float* b1    = (const float*)d_in[13];  // [512]
    const float* W2    = (const float*)d_in[14];  // [512,128]
    const float* b2    = (const float*)d_in[15];  // [128]
    float* out = (float*)d_out;                   // [16,1536,128]

    char* w = (char*)d_ws;
    auto alloc = [&](size_t bytes) -> char* {
        char* p = w; w += (bytes + 255) & ~(size_t)255; return p;
    };
    bf16_t* avb   = (bf16_t*)alloc(8192ull * 128 * 2);   // embedded atoms, bf16
    bf16_t* Wg2T  = (bf16_t*)alloc(256ull * 128 * 2);    // [h*64+f][c]
    bf16_t* WoutT = (bf16_t*)alloc(128ull * 256 * 2);    // [o][i]
    bf16_t* WcT   = (bf16_t*)alloc(128ull * 128 * 2);
    bf16_t* W1T   = (bf16_t*)alloc(512ull * 1152 * 2);
    bf16_t* W2T   = (bf16_t*)alloc(128ull * 512 * 2);
    float*  Wh    = (float*) alloc(8192ull * 256 * 4);   // [b*512+n][h*64+f]
    float*  s1    = (float*) alloc(32768ull * 4);
    float*  s2    = (float*) alloc(32768ull * 4);
    bf16_t* WhT   = (bf16_t*)alloc(64ull * 64 * 512 * 2);   // per (b,h): [f][m]
    bf16_t* att1  = (bf16_t*)alloc(64ull * 512 * 512 * 2);
    bf16_t* multi = (bf16_t*)alloc(8192ull * 256 * 2);      // elu(hp), [b*512+n][h*64+f]
    float*  Wh2   = (float*) alloc(8192ull * 128 * 4);
    float*  s1b   = (float*) alloc(8192ull * 4);
    float*  s2b   = (float*) alloc(8192ull * 4);
    bf16_t* Wh2T  = (bf16_t*)alloc(16ull * 128 * 512 * 2);  // per b: [o][m]
    bf16_t* att2  = (bf16_t*)alloc(16ull * 512 * 512 * 2);
    bf16_t* outb  = (bf16_t*)alloc(8192ull * 128 * 2);      // elu(hp2)
    bf16_t* protb = (bf16_t*)alloc(16384ull * 1152 * 2);
    bf16_t* hidb  = (bf16_t*)alloc(16384ull * 512 * 2);
    (void)ws_size; (void)in_sizes; (void)n_in; (void)out_size;

    auto blocks = [](long total, int bs) { return dim3((unsigned)((total + bs - 1) / bs)); };

    // --- Stage 0: weight pre-transposes + input conversions (independent) ---
    transpose_f32_bf16<<<blocks(4L*128*64, 256), 256, 0, stream>>>(
        Wg2T, Wgat, 128, 64, 64, 1, 8192, 0, 4L*128*64);          // per-head W_gat^T
    transpose_f32_bf16<<<blocks(256L*128, 256), 256, 0, stream>>>(
        WoutT, Wout, 256, 128, 128, 1, 0, 0, 256L*128);
    transpose_f32_bf16<<<blocks(128L*128, 256), 256, 0, stream>>>(
        WcT, Wc, 128, 128, 128, 1, 0, 0, 128L*128);
    transpose_f32_bf16<<<blocks(1152L*512, 256), 256, 0, stream>>>(
        W1T, W1, 1152, 512, 512, 1, 0, 0, 1152L*512);
    transpose_f32_bf16<<<blocks(512L*128, 256), 256, 0, stream>>>(
        W2T, W2, 512, 128, 128, 1, 0, 0, 512L*128);
    embed_kernel<<<blocks(8192L*128, 256), 256, 0, stream>>>(avb, atoms, emb, 8192L*128);
    cvt_bf16_kernel<<<blocks(16384L*1152, 256), 256, 0, stream>>>(protb, prot, 16384L*1152);

    // --- GEMM1: Wh = av @ W_gat (all heads)   [8192,128]x[128,256] ---
    gemm_bf16_wmma<0, float><<<dim3(256, 2, 1), 128, 0, stream>>>(
        avb, 128, 0, Wg2T, 128, 0, Wh, 256, 1, 0, 0, nullptr, 8192, 256, 128);

    // --- scores + V transpose for GAT1 ---
    gat1_scores_kernel<<<blocks(32768, 256), 256, 0, stream>>>(s1, s2, Wh, agat);
    transpose_f32_bf16<<<blocks(64L*64*512, 256), 256, 0, stream>>>(
        WhT, Wh, 512, 64, 256, 4, 512L*256, 64, 64L*64*512);      // WhT[z][f][m]

    // --- masked softmax (B*H rows of 512) ---
    gat_softmax_kernel<<<dim3(32768 / 8), 256, 0, stream>>>(att1, s1, s2, adj, 4, 32768);

    // --- GEMM2: multi = elu(att1 @ Wh)  per (b,h): [512,512]x[512,64] -> bf16 ---
    gemm_bf16_wmma<1, bf16_t><<<dim3(16, 1, 64), 128, 0, stream>>>(
        att1, 512, 512L*512, WhT, 512, 64L*512,
        multi, 256, 4, 512L*256, 64, nullptr, 512, 64, 512);

    // --- GEMM3: Wh2 = multi @ W_out   [8192,256]x[256,128] ---
    gemm_bf16_wmma<0, float><<<dim3(256, 1, 1), 128, 0, stream>>>(
        multi, 256, 0, WoutT, 256, 0, Wh2, 128, 1, 0, 0, nullptr, 8192, 128, 256);

    // --- scores + V transpose for GAT2 ---
    gat2_scores_kernel<<<blocks(8192, 256), 256, 0, stream>>>(s1b, s2b, Wh2, aout);
    transpose_f32_bf16<<<blocks(16L*128*512, 256), 256, 0, stream>>>(
        Wh2T, Wh2, 512, 128, 128, 1, 512L*128, 0, 16L*128*512);

    gat_softmax_kernel<<<dim3(8192 / 8), 256, 0, stream>>>(att2, s1b, s2b, adj, 1, 8192);

    // --- GEMM4: outb = elu(att2 @ Wh2)  per b: [512,512]x[512,128] -> bf16 ---
    gemm_bf16_wmma<1, bf16_t><<<dim3(16, 1, 16), 128, 0, stream>>>(
        att2, 512, 512L*512, Wh2T, 512, 128L*512,
        outb, 128, 1, 512L*128, 0, nullptr, 512, 128, 512);

    // --- GEMM5: atoms_vector = leaky(outb @ Wc + bc) -> d_out rows b*1536+n ---
    gemm_bf16_wmma<3, float><<<dim3(16, 1, 16), 128, 0, stream>>>(
        outb, 128, 512L*128, WcT, 128, 0,
        out, 128, 1, 1536L*128, 0, bc, 512, 128, 128);

    // --- GEMM6: hid = relu(prot @ W1 + b1)   [16384,1152]x[1152,512] -> bf16 ---
    gemm_bf16_wmma<2, bf16_t><<<dim3(512, 4, 1), 128, 0, stream>>>(
        protb, 1152, 0, W1T, 1152, 0, hidb, 512, 1, 0, 0, b1, 16384, 512, 1152);

    // --- GEMM7: amino = leaky(hid @ W2 + b2) -> d_out rows b*1536+512+l ---
    gemm_bf16_wmma<3, float><<<dim3(32, 1, 16), 128, 0, stream>>>(
        hidb, 512, 1024L*512, W2T, 512, 0,
        out + 512L*128, 128, 1, 1536L*128, 0, b2, 1024, 128, 512);
}
